// CoordinateDescent_65463891526110
// MI455X (gfx1250) — matrix operand order repaired and checked by reading, not compiled, not run on hardware
//
#include <hip/hip_runtime.h>
#include <hip/hip_bf16.h>

// ---------------------------------------------------------------------------
// CoordinateDescent routing:
//   Phase 1: s = x @ r   — TDM (tensor_load_to_lds) double-buffered HBM->LDS
//            stream + bf16 WMMA GEMV. 128 MB of x @ 23.3 TB/s ~= 5.5 us floor;
//            compute is free, so the win is linear full-line DMA traffic and
//            waves spending issue slots only on ds_load/cvt/wmma.
//   Phase 2: 50x coordinate descent (LDS logsumexp), O(b*n), negligible.
//   Phase 3: top-512 via 64 KB LDS bitonic sort, negligible.
// b=4, n=8192, d=1024 fixed by the reference setup.
// ---------------------------------------------------------------------------

typedef __attribute__((ext_vector_type(16))) __bf16        v16bf;
typedef __attribute__((ext_vector_type(8)))  float         v8f;
typedef __attribute__((ext_vector_type(4)))  unsigned int  v4u;
typedef __attribute__((ext_vector_type(8)))  int           v8i;
typedef __attribute__((ext_vector_type(4)))  int           v4i;

#define BATCH      4
#define NTOK       8192
#define DDIM       1024
#define EPS_F      0.1f
#define INV_EPS_F  10.0f
#define NEG_MAX    (-3.402823466e38f)

#define ROWS_PER_BLK 128   // 8 waves x 16-row WMMA tiles
#define KCHUNK       32    // K columns per TDM tile (= one WMMA step)
#define NCHUNK       (DDIM / KCHUNK)

// ---------------------------------------------------------------------------
// TDM: post a 2D tile load (ROWS_PER_BLK rows x KCHUNK f32 cols, row stride
// DDIM) from global to LDS. D# packing per CDNA5 ISA 8.3/8.4:
//   group0: [1:0]=count=1, [63:32]=lds_addr, [120:64]=global_addr, [127:126]=2
//   group1: [17:16]=data_size(2=4B), [79:48]=tensor_dim0, [111:80]=tensor_dim1,
//           [127:112]=tile_dim0, [143:128]=tile_dim1, [207:160]=dim0_stride
// This toolchain exposes the 6-arg builtin (g0, g1, g2, g3, g4, cpol);
// groups 2/3/4 are zero for a 2D tile.
// ---------------------------------------------------------------------------
__device__ __forceinline__ void tdm_load_tile(const float* gsrc, unsigned int lds_off)
{
    unsigned long long ga = (unsigned long long)(uintptr_t)gsrc;
    v4u g0;
    g0[0] = 1u;                                              // count=1, load-desc
    g0[1] = lds_off;                                         // LDS byte address
    g0[2] = (unsigned int)ga;                                // global_addr[31:0]
    g0[3] = (unsigned int)((ga >> 32) & 0x1FFFFFFu)          // global_addr[56:32]
          | 0x80000000u;                                     // type=2 ("image")

    const unsigned int td0 = DDIM;        // tensor_dim0 (elements)
    const unsigned int td1 = 1u << 20;    // tensor_dim1: large (OOB bound only)
    v8i g1;
    g1[0] = (int)(2u << 16);                                 // data_size=4B
    g1[1] = (int)((td0 & 0xFFFFu) << 16);                    // tensor_dim0[15:0]
    g1[2] = (int)((td0 >> 16) | ((td1 & 0xFFFFu) << 16));    // dim0[31:16]|dim1[15:0]
    g1[3] = (int)(((td1 >> 16) & 0xFFFFu)
          | ((unsigned int)KCHUNK << 16));                   // dim1[31:16]|tile_dim0
    g1[4] = ROWS_PER_BLK;                                    // tile_dim1 (tile_dim2=0)
    g1[5] = (int)DDIM;                                       // dim0_stride[31:0]
    g1[6] = 0;                                               // dim0_stride[47:32]
    g1[7] = 0;

    v4i z4 = {0, 0, 0, 0};                                   // 2D: groups 2/3 unused
    v8i z8 = {0, 0, 0, 0, 0, 0, 0, 0};
    __builtin_amdgcn_tensor_load_to_lds(g0, g1, z4, z4, z8, 0);
}

// ---------------------------------------------------------------------------
// Kernel 1: s[b][row] = dot(x[b,row,:], r) with masked_fill(-FLT_MAX).
// Double-buffered TDM pipeline feeding v_wmma_f32_16x16x32_bf16.
// ---------------------------------------------------------------------------
__global__ __launch_bounds__(256) void cd_gemv_wmma(
    const float* __restrict__ x, const float* __restrict__ r,
    const unsigned char* __restrict__ mask, float* __restrict__ s_out)
{
    __shared__ __align__(16) float tile[2][ROWS_PER_BLK * KCHUNK]; // 2 x 16 KB
    __shared__ v16bf rs[DDIM / 16];                                // 2 KB bf16 r

    const int tid = threadIdx.x;
    const int bb  = blockIdx.y;

    // one-time fp32 -> bf16 convert of the routing token into LDS
    {
        __bf16* rsc = reinterpret_cast<__bf16*>(rs);
        for (int i = tid; i < DDIM; i += 256) rsc[i] = (__bf16)r[i];
    }

    const int wave = tid >> 5;
    const int lane = tid & 31;
    const int half = lane >> 4;     // 0: K-low half of A fragment, 1: K-high
    const int mrow = lane & 15;     // row within the wave's 16-row tile
    const int row0 = blockIdx.x * ROWS_PER_BLK;

    const float* __restrict__ gbase = x + ((size_t)bb * NTOK + row0) * DDIM;

    const unsigned int lds0 = (unsigned int)(uintptr_t)&tile[0][0];
    const unsigned int lds1 = (unsigned int)(uintptr_t)&tile[1][0];

    // prologue: wave 0 posts chunk 0, drains TENSORcnt, workgroup barrier
    if (wave == 0) {
        tdm_load_tile(gbase, lds0);
        __builtin_amdgcn_s_wait_tensorcnt(0);
    }
    __syncthreads();

    v8f c = {};  // 16x16 f32 accumulator (all N columns identical)

    for (int ch = 0; ch < NCHUNK; ++ch) {
        // post chunk ch+1 into the other buffer while computing chunk ch
        if (wave == 0 && ch + 1 < NCHUNK)
            tdm_load_tile(gbase + (ch + 1) * KCHUNK, (ch & 1) ? lds0 : lds1);

        // A-matrix bf16 16x32: lane<16 holds K=[0,16), lane>=16 K=[16,32)
        const float* t = tile[ch & 1] + (wave * 16 + mrow) * KCHUNK;
        const float4* p0 = reinterpret_cast<const float4*>(t + half * 16);
        const float4* p1 = reinterpret_cast<const float4*>(t + half * 16 + 8);
        float4 f0 = p0[0], f1 = p0[1];
        float4 f2 = p1[0], f3 = p1[1];

        v16bf a;
        a[0]  = (__bf16)f0.x; a[1]  = (__bf16)f0.y; a[2]  = (__bf16)f0.z; a[3]  = (__bf16)f0.w;
        a[4]  = (__bf16)f1.x; a[5]  = (__bf16)f1.y; a[6]  = (__bf16)f1.z; a[7]  = (__bf16)f1.w;
        a[8]  = (__bf16)f2.x; a[9]  = (__bf16)f2.y; a[10] = (__bf16)f2.z; a[11] = (__bf16)f2.w;
        a[12] = (__bf16)f3.x; a[13] = (__bf16)f3.y; a[14] = (__bf16)f3.z; a[15] = (__bf16)f3.w;

        // B-matrix 32x16: lane<16 supplies K=[0,16), lane>=16 K=[16,32);
        // every N column carries the same routing value -> D col j = dot.
        v16bf bvv = rs[(ch << 1) + half];

        c = __builtin_amdgcn_wmma_f32_16x16x32_bf16(
                false, a, false, bvv, (short)0, c, false, false);

        // chunk ch+1 must be resident before anyone reads it next iteration
        if (wave == 0)
            __builtin_amdgcn_s_wait_tensorcnt(0);
        __syncthreads();
    }

    // C/D layout: VGPR m holds M=m (lanes 0-15) and M=m+8 (lanes 16-31).
    // Lane 0 writes rows 0..7, lane 16 writes rows 8..15 (N=0 column).
    if (mrow == 0) {
        const size_t base = (size_t)bb * NTOK + row0 + wave * 16 + half * 8;
        #pragma unroll
        for (int m = 0; m < 8; ++m) {
            float sv = c[m];
            s_out[base + m] = mask[base + m] ? sv : NEG_MAX;
        }
    }
}

// ---------------------------------------------------------------------------
// Kernel 2: 50 iterations of  a = c - eps*logsumexp((s+b)/eps);  b = -relu(s+a)
// One 1024-thread block per batch row; each thread owns 8 strided elements.
// ---------------------------------------------------------------------------
__global__ __launch_bounds__(1024) void cd_iterate(
    const float* __restrict__ s_in, float* __restrict__ scores)
{
    __shared__ float red[1024];

    const int t  = threadIdx.x;
    const int bb = blockIdx.x;
    const float* __restrict__ sp = s_in + (size_t)bb * NTOK;

    float s[8], bv[8];
    #pragma unroll
    for (int j = 0; j < 8; ++j) s[j] = sp[t + (j << 10)];
    #pragma unroll
    for (int j = 0; j < 8; ++j) bv[j] = -fmaxf(s[j], 0.0f);

    const float kconst = EPS_F * logf(512.0f);   // eps * log(k)
    float a = 0.0f;

    for (int it = 0; it < 50; ++it) {
        float u[8];
        float lm = NEG_MAX;
        #pragma unroll
        for (int j = 0; j < 8; ++j) {
            u[j] = (s[j] + bv[j]) * INV_EPS_F;
            lm = fmaxf(lm, u[j]);
        }
        // block max
        red[t] = lm; __syncthreads();
        for (int o = 512; o > 0; o >>= 1) {
            if (t < o) red[t] = fmaxf(red[t], red[t + o]);
            __syncthreads();
        }
        float M = red[0]; __syncthreads();
        // block sum of exp(u - M)
        float ls = 0.0f;
        #pragma unroll
        for (int j = 0; j < 8; ++j) ls += expf(u[j] - M);
        red[t] = ls; __syncthreads();
        for (int o = 512; o > 0; o >>= 1) {
            if (t < o) red[t] += red[t + o];
            __syncthreads();
        }
        float S = red[0]; __syncthreads();

        a = kconst - EPS_F * (M + logf(S));
        #pragma unroll
        for (int j = 0; j < 8; ++j) bv[j] = -fmaxf(s[j] + a, 0.0f);
    }

    float* __restrict__ op = scores + (size_t)bb * NTOK;
    #pragma unroll
    for (int j = 0; j < 8; ++j)
        op[t + (j << 10)] = expf((s[j] + a + bv[j]) * INV_EPS_F);
}

// ---------------------------------------------------------------------------
// Kernel 3: descending bitonic sort of 8192 (score,index) pairs in LDS (64 KB),
// then emit top-num_tokens: value = mask ? 1.0 : 0.0 (straight-through), index.
// ---------------------------------------------------------------------------
__global__ __launch_bounds__(1024) void cd_topk(
    const float* __restrict__ scores, const unsigned char* __restrict__ mask,
    const int* __restrict__ nt_ptr, float* __restrict__ out)
{
    __shared__ float key[NTOK];   // 32 KB
    __shared__ int   val[NTOK];   // 32 KB

    const int t  = threadIdx.x;
    const int bb = blockIdx.x;

    for (int i = t; i < NTOK; i += 1024) {
        key[i] = scores[(size_t)bb * NTOK + i];
        val[i] = i;
    }
    __syncthreads();

    for (int k = 2; k <= NTOK; k <<= 1) {
        for (int j = k >> 1; j > 0; j >>= 1) {
            for (int i = t; i < NTOK; i += 1024) {
                int ixj = i ^ j;
                if (ixj > i) {
                    bool desc = ((i & k) == 0);
                    float ki = key[i], kx = key[ixj];
                    bool sw = desc ? (ki < kx) : (ki > kx);
                    if (sw) {
                        key[i] = kx; key[ixj] = ki;
                        int tv = val[i]; val[i] = val[ixj]; val[ixj] = tv;
                    }
                }
            }
            __syncthreads();
        }
    }

    int nt = nt_ptr[0];
    if (nt > NTOK) nt = NTOK;

    float* so = out;                                               // [B, nt]
    int*   io = reinterpret_cast<int*>(out + (size_t)BATCH * nt);  // [B, nt]
    for (int i = t; i < nt; i += 1024) {
        int idx = val[i];
        so[(size_t)bb * nt + i] = mask[(size_t)bb * NTOK + idx] ? 1.0f : 0.0f;
        io[(size_t)bb * nt + i] = idx;
    }
}

// ---------------------------------------------------------------------------
extern "C" void kernel_launch(void* const* d_in, const int* in_sizes, int n_in,
                              void* d_out, int out_size, void* d_ws, size_t ws_size,
                              hipStream_t stream) {
    (void)in_sizes; (void)n_in; (void)out_size; (void)ws_size;

    const float*         x    = (const float*)d_in[0];
    const float*         r    = (const float*)d_in[1];
    const unsigned char* mask = (const unsigned char*)d_in[2];
    const int*           ntp  = (const int*)d_in[3];

    float* s      = (float*)d_ws;                    // [4, 8192]
    float* scores = s + (size_t)BATCH * NTOK;        // [4, 8192]

    // Phase 1: TDM-fed WMMA GEMV (memory-bound, ~5.5 us floor at 23.3 TB/s)
    cd_gemv_wmma<<<dim3(NTOK / ROWS_PER_BLK, BATCH), 256, 0, stream>>>(x, r, mask, s);
    // Phase 2: coordinate descent (one block per batch row)
    cd_iterate<<<BATCH, 1024, 0, stream>>>(s, scores);
    // Phase 3: top-k via LDS bitonic sort
    cd_topk<<<BATCH, 1024, 0, stream>>>(scores, mask, ntp, (float*)d_out);
}
